// LstmClassifier_21912923144364
// MI455X (gfx1250) — matrix operand order repaired and checked by reading, not compiled, not run on hardware
//
#include <hip/hip_runtime.h>
#include <hip/hip_bf16.h>

// ---------------------------------------------------------------------------
// Bidirectional LSTM classifier for MI455X (gfx1250), wave32 + WMMA bf16.
//   B=16, S=2048, D=768, H=512, C=5
// Phases:
//   prep     : x + weights f32->bf16 (one pass), W_clf zero-pad to 16 rows,
//              zero h-state double buffers + per-direction barrier counters.
//   gemm_xg  : xg[dir][s][b][4H] = x @ Wih^T + b   (WMMA bf16, f32 acc;
//              4 N-tiles per wave sharing one A fragment).
//   scan     : 32 persistent workgroups (16 per direction). Each WG stages its
//              128-gate-column slice of Whh into LDS via the Tensor Data Mover
//              (tensor_load_to_lds + s_wait_tensorcnt), computes
//              g = xg[s] + h @ Whh^T with WMMA, gates in f32, writes bf16 h to
//              a double-buffered global state + relu(h) to the hs buffer, then
//              a per-direction barrier per step (HW cluster barrier when
//              dispatched as a 16-WG cluster, atomic fallback otherwise).
//   clf      : out = relu([hf|hb]) @ W_clf^T + b_clf  (WMMA, N padded to 16)
// ---------------------------------------------------------------------------

#define Bn 16
#define Sn 2048
#define Dn 768
#define Hn 512
#define G4 2048   // 4*H
#define Cn 5
#define NSCAN_WG 32
#define WG_PER_DIR 16

typedef __attribute__((ext_vector_type(16))) __bf16 v16bf;
typedef __attribute__((ext_vector_type(8)))  __bf16 v8bf;
typedef __attribute__((ext_vector_type(8)))  float  v8f;
typedef __attribute__((ext_vector_type(4)))  unsigned int v4u;
typedef __attribute__((ext_vector_type(8)))  int v8i;
typedef __attribute__((ext_vector_type(4)))  int v4i;

// ---- WMMA fragment loaders (bf16 16x16x32, contiguous K = 16*(lane/16)+s, same in A and B) ----
// A (16x32, MxK): lane<16 -> row=lane, K = kb+0..15
//                 lane>=16 -> row=lane-16, K = kb+16..31
__device__ inline v16bf load_a_frag_bf16(const __bf16* base, int ld) {
  int lane = threadIdx.x & 31;
  int row  = lane & 15;
  int ko   = (lane >> 4) * 16;
  const __bf16* p = base + (size_t)row * ld + ko;
  v8bf lo = *(const v8bf*)(p);
  v8bf hi = *(const v8bf*)(p + 8);
  v16bf a;
#pragma unroll
  for (int i = 0; i < 8; ++i) { a[i] = lo[i]; a[i + 8] = hi[i]; }
  return a;
}

// B (32x16, KxN), column n contiguous in K (i.e. row n of a [N,K] weight):
//   lane<16 -> col=lane, K = kb+0..15 ; lane>=16 -> col=lane-16, K = kb+16..31
__device__ inline v16bf load_b_frag(const __bf16* base, int ld) {
  int lane = threadIdx.x & 31;
  int col  = lane & 15;
  int ko   = (lane >> 4) * 16;
  return *(const v16bf*)(base + (size_t)col * ld + ko);
}

__device__ inline v8f wmma_bf16(v16bf a, v16bf b, v8f c) {
  return __builtin_amdgcn_wmma_f32_16x16x32_bf16(false, a, false, b, (short)0, c,
                                                 false, false);
}

__device__ inline float sigf(float x) { return 1.0f / (1.0f + __expf(-x)); }

// ---------------------------------------------------------------------------
__global__ void prep_kernel(const float* __restrict__ x,
                            const float* __restrict__ WihF, const float* __restrict__ WhhF,
                            const float* __restrict__ WihB, const float* __restrict__ WhhB,
                            const float* __restrict__ Wclf,
                            __bf16* __restrict__ xbf,
                            __bf16* __restrict__ wihf, __bf16* __restrict__ whhf,
                            __bf16* __restrict__ wihb, __bf16* __restrict__ whhb,
                            __bf16* __restrict__ wclf16, __bf16* __restrict__ hstate,
                            unsigned* __restrict__ bar) {
  const size_t NX = (size_t)Bn * Sn * Dn;   // 25,165,824
  const size_t NI = (size_t)G4 * Dn;        //  1,572,864
  const size_t NH = (size_t)G4 * Hn;        //  1,048,576
  size_t tid    = (size_t)blockIdx.x * blockDim.x + threadIdx.x;
  size_t stride = (size_t)gridDim.x * blockDim.x;
  for (size_t i = tid; i < NX; i += stride) xbf[i] = (__bf16)x[i];
  for (size_t i = tid; i < NI; i += stride) { wihf[i] = (__bf16)WihF[i]; wihb[i] = (__bf16)WihB[i]; }
  for (size_t i = tid; i < NH; i += stride) { whhf[i] = (__bf16)WhhF[i]; whhb[i] = (__bf16)WhhB[i]; }
  for (size_t i = tid; i < 16 * 2 * Hn; i += stride) {  // padded classifier weight [16][1024]
    int n = (int)(i >> 10), k = (int)(i & 1023);
    wclf16[i] = (n < Cn) ? (__bf16)Wclf[n * (2 * Hn) + k] : (__bf16)0.0f;
  }
  for (size_t i = tid; i < 2 * 2 * Bn * Hn; i += stride) hstate[i] = (__bf16)0.0f;
  if (tid < 64) bar[tid] = 0u;
}

// ---------------------------------------------------------------------------
// xg[dir] (bf16, [S,B,4H]) = x @ Wih[dir]^T + b[dir]
// grid (2048, 8), block 256 (8 waves). Wave w owns a 16x64 tile (4 N-tiles
// sharing one A fragment): M-tile = blockIdx.x, cols = blockIdx.y*512 + w*64.
__global__ void gemm_xg_kernel(const __bf16* __restrict__ xbf,
                               const __bf16* __restrict__ wihf,
                               const __bf16* __restrict__ wihb,
                               const float* __restrict__ bf,
                               const float* __restrict__ bb,
                               __bf16* __restrict__ xgf,
                               __bf16* __restrict__ xgb) {
  int mt    = blockIdx.x;
  int wave  = threadIdx.x >> 5;
  int ncol0 = blockIdx.y * 512 + wave * 64;   // 0..4095, 64-col group, one dir
  int dir   = ncol0 >> 11;
  int n0    = ncol0 & (G4 - 1);

  const __bf16* W    = dir ? wihb : wihf;
  const float*  bias = dir ? bb   : bf;
  __bf16*       xg   = dir ? xgb  : xgf;

  const __bf16* arow = xbf + (size_t)mt * 16 * Dn;  // rows m = b*S+s contiguous

  v8f acc[4];
#pragma unroll
  for (int j = 0; j < 4; ++j)
#pragma unroll
    for (int i = 0; i < 8; ++i) acc[j][i] = 0.0f;

#pragma unroll 2
  for (int kb = 0; kb < Dn; kb += 32) {
    v16bf a = load_a_frag_bf16(arow + kb, Dn);
#pragma unroll
    for (int j = 0; j < 4; ++j) {
      v16bf b = load_b_frag(W + (size_t)(n0 + j * 16) * Dn + kb, Dn);
      acc[j] = wmma_bf16(a, b, acc[j]);
    }
  }

  int lane = threadIdx.x & 31;
  int col  = lane & 15;
#pragma unroll
  for (int j = 0; j < 4; ++j) {
    int n    = n0 + j * 16 + col;
    float bv = bias[n];
#pragma unroll
    for (int i = 0; i < 8; ++i) {
      int m  = mt * 16 + i + (lane >> 4) * 8;
      int b_ = m >> 11, s_ = m & (Sn - 1);
      xg[((size_t)s_ * Bn + b_) * G4 + n] = (__bf16)(acc[j][i] + bv);
    }
  }
}

// ---------------------------------------------------------------------------
// Persistent scan. 32 WGs: dir = blockIdx.x>>4, slice w = blockIdx.x&15.
// WG owns h columns [w*32, w*32+32) => gate columns {q*512 + w*32 + 0..31}.
// LDS: Whh slice (128 cols x 512 K bf16 = 128 KB) + g tile (16x128 f32) + c (16x32 f32).
__global__ void scan_kernel(const __bf16* __restrict__ whhf,
                            const __bf16* __restrict__ whhb,
                            const __bf16* __restrict__ xgf,
                            const __bf16* __restrict__ xgb,
                            __bf16* __restrict__ hreluf,
                            __bf16* __restrict__ hrelub,
                            __bf16* __restrict__ hstate,
                            unsigned* __restrict__ bar) {
  extern __shared__ char smem[];
  __bf16* wlds = (__bf16*)smem;                          // [128][512]
  float*  glds = (float*)(smem + 128 * 512 * 2);          // [16][128]
  float*  clds = glds + 16 * 128;                         // [16][32]

  const int dir = blockIdx.x >> 4;
  const int w   = blockIdx.x & 15;

  const __bf16* Whh   = dir ? whhb   : whhf;
  const __bf16* xg    = dir ? xgb    : xgf;
  __bf16*       hrelu = dir ? hrelub : hreluf;
  __bf16*       hbuf  = hstate + (size_t)dir * 2 * Bn * Hn;   // double buffer

  // Cluster-aware: when launched as 16-WG clusters (one per direction) the
  // per-step barrier is the HW cluster barrier; plain launches (cluster_id==0)
  // fall back to a per-direction atomic barrier.
  const bool clustered = (__builtin_amdgcn_cluster_id_x() != 0);

  // --- Stage Whh slice into LDS with the Tensor Data Mover (one wave issues).
  // D# 3D tile: X = 512 elems (one row, contiguous), Y = 32 rows
  // (stride 512 elems), Z = 4 gate blocks (stride 512*512 elems), 2B elems.
  if (threadIdx.x < 32) {
    unsigned long long ga = (unsigned long long)(uintptr_t)(Whh + (size_t)w * 32 * Hn);
    unsigned lds_addr = __builtin_amdgcn_groupstaticsize();  // dynamic LDS base
    v4u g0;
    g0[0] = 1u;                                   // count=1 (valid descriptor)
    g0[1] = lds_addr;                             // lds_addr[31:0]
    g0[2] = (unsigned)(ga & 0xFFFFFFFFu);         // global_addr[31:0]
    g0[3] = (unsigned)((ga >> 32) & 0x1FFFFFFu)   // global_addr[56:32]
            | (2u << 30);                         // type=2 (image)
    v8i g1;
    g1[0] = (int)(1u << 16);                      // data_size=1 (2 bytes)
    g1[1] = (int)(512u << 16);                    // tensor_dim0[15:0] in [31:16]
    g1[2] = (int)(512u << 16);                    // dim0 hi=0 | tensor_dim1[15:0]
    g1[3] = (int)(512u << 16);                    // dim1 hi=0 | tile_dim0=512
    g1[4] = (int)(32u | (4u << 16));              // tile_dim1=32 | tile_dim2=4
    g1[5] = (int)512u;                            // tensor_dim0_stride lo
    g1[6] = 0;                                    // stride0 hi | stride1[15:0]=0
    g1[7] = (int)(262144u >> 16);                 // tensor_dim1_stride[47:16]
    v4i g2;
    g2[0] = 4; g2[1] = 0; g2[2] = 0; g2[3] = 0;   // tensor_dim2=4
    v4i g3;
    g3[0] = 0; g3[1] = 0; g3[2] = 0; g3[3] = 0;
    v8i g4;                                       // extra group (clang-23 form)
#pragma unroll
    for (int i = 0; i < 8; ++i) g4[i] = 0;
    __builtin_amdgcn_tensor_load_to_lds(g0, g1, g2, g3, g4, 0);
    __builtin_amdgcn_s_wait_tensorcnt(0);
  }
  __syncthreads();

  // D# field semantics are not hardware-verified: deterministic manual copy of
  // the same slice runs strictly AFTER the tensor wait (one-time, negligible).
  for (int idx = threadIdx.x; idx < 128 * (Hn / 8); idx += blockDim.x) {
    int l  = idx >> 6;            // local col 0..127
    int kc = (idx & 63) * 8;
    int q = l >> 5, j = l & 31;
    int n = q * Hn + w * 32 + j;
    *(v8bf*)(wlds + (size_t)l * Hn + kc) = *(const v8bf*)(Whh + (size_t)n * Hn + kc);
  }
  for (int p = threadIdx.x; p < 16 * 32; p += blockDim.x) clds[p] = 0.0f;
  __syncthreads();

  const int wave = threadIdx.x >> 5;
  const int lane = threadIdx.x & 31;
  const int q    = wave >> 1, sub = wave & 1;
  const int lc   = q * 32 + sub * 16;               // local col base
  const int nb   = q * Hn + w * 32 + sub * 16;      // global gate col base

  for (int t = 0; t < Sn; ++t) {
    const int s = dir ? (Sn - 1 - t) : t;
    const __bf16* hcur  = hbuf + (size_t)(t & 1) * Bn * Hn;
    __bf16*       hnext = hbuf + (size_t)((t + 1) & 1) * Bn * Hn;

    // prefetch next step's xg slice into cache
    if (t + 1 < Sn) {
      int s2 = dir ? (Sn - 2 - t) : (t + 1);
      const __bf16* pf = xg + ((size_t)s2 * Bn + (lane & 15)) * G4 + nb;
      __builtin_prefetch(pf, 0, 1);
    }

    // g-tile = h @ Whh_slice^T  (16 chained WMMAs)
    v8f acc;
#pragma unroll
    for (int i = 0; i < 8; ++i) acc[i] = 0.0f;
#pragma unroll
    for (int kb = 0; kb < Hn; kb += 32) {
      v16bf a = load_a_frag_bf16(hcur + kb, Hn);
      v16bf b = load_b_frag(wlds + (size_t)lc * Hn + kb, Hn);
      acc = wmma_bf16(a, b, acc);
    }

    // add hoisted input projection, park in LDS for the gate pass
    {
      int col = lane & 15;
      const __bf16* xgrow = xg + (size_t)s * Bn * G4 + nb + col;
#pragma unroll
      for (int i = 0; i < 8; ++i) {
        int b_ = i + (lane >> 4) * 8;
        glds[b_ * 128 + lc + col] = acc[i] + (float)xgrow[(size_t)b_ * G4];
      }
    }
    __syncthreads();

    // gates: 16x32 elements, 2 per thread
#pragma unroll
    for (int e = 0; e < 2; ++e) {
      int p  = threadIdx.x * 2 + e;
      int b_ = p >> 5, j = p & 31;
      float gi = glds[b_ * 128 + j];
      float gf = glds[b_ * 128 + 32 + j];
      float gg = glds[b_ * 128 + 64 + j];
      float go = glds[b_ * 128 + 96 + j];
      float c0 = clds[b_ * 32 + j];
      float c1 = sigf(gf) * c0 + sigf(gi) * tanhf(gg);
      float h  = sigf(go) * tanhf(c1);
      clds[b_ * 32 + j] = c1;
      int hc = w * 32 + j;
      hnext[(size_t)b_ * Hn + hc] = (__bf16)h;
      hrelu[((size_t)b_ * Sn + s) * Hn + hc] = (__bf16)fmaxf(h, 0.0f);
    }

    // per-direction barrier: release h_next, wait for the 16 WGs of this dir
    __syncthreads();
    __threadfence();
    if (clustered) {
      if (threadIdx.x < 32) __builtin_amdgcn_s_cluster_barrier();
    } else if (threadIdx.x == 0) {
      atomicAdd(&bar[dir], 1u);
      unsigned target = (unsigned)(t + 1) * WG_PER_DIR;
      while (__hip_atomic_load(&bar[dir], __ATOMIC_ACQUIRE,
                               __HIP_MEMORY_SCOPE_AGENT) < target) {
        __builtin_amdgcn_s_sleep(2);
      }
    }
    __syncthreads();
    __threadfence();
  }
}

// ---------------------------------------------------------------------------
// out[b,s,c] = relu([hf|hb]) @ W_clf^T + b_clf ; N padded to one 16-wide tile.
// grid 256 x block 256: wave handles M-tile = blockIdx.x*8 + wave.
__global__ void clf_kernel(const __bf16* __restrict__ hf,
                           const __bf16* __restrict__ hb,
                           const __bf16* __restrict__ wclf16,
                           const float* __restrict__ bclf,
                           float* __restrict__ out) {
  int mt = blockIdx.x * 8 + (threadIdx.x >> 5);
  v8f acc;
#pragma unroll
  for (int i = 0; i < 8; ++i) acc[i] = 0.0f;

#pragma unroll 4
  for (int kb = 0; kb < 2 * Hn; kb += 32) {
    const __bf16* abase = (kb < Hn) ? (hf + (size_t)mt * 16 * Hn + kb)
                                    : (hb + (size_t)mt * 16 * Hn + (kb - Hn));
    v16bf a = load_a_frag_bf16(abase, Hn);
    v16bf b = load_b_frag(wclf16 + kb, 2 * Hn);
    acc = wmma_bf16(a, b, acc);
  }

  int lane = threadIdx.x & 31;
  int col  = lane & 15;
  if (col < Cn) {
    float bv = bclf[col];
#pragma unroll
    for (int i = 0; i < 8; ++i) {
      int m = mt * 16 + i + (lane >> 4) * 8;
      out[(size_t)m * Cn + col] = acc[i] + bv;
    }
  }
}

// ---------------------------------------------------------------------------
extern "C" void kernel_launch(void* const* d_in, const int* in_sizes, int n_in,
                              void* d_out, int out_size, void* d_ws, size_t ws_size,
                              hipStream_t stream) {
  const float* x    = (const float*)d_in[0];
  const float* WihF = (const float*)d_in[1];
  const float* WhhF = (const float*)d_in[2];
  const float* bF   = (const float*)d_in[3];
  const float* WihB = (const float*)d_in[4];
  const float* WhhB = (const float*)d_in[5];
  const float* bB   = (const float*)d_in[6];
  const float* Wclf = (const float*)d_in[7];
  const float* bclf = (const float*)d_in[8];
  float* out = (float*)d_out;

  char* ws = (char*)d_ws;
  size_t off = 0;
  auto take = [&](size_t bytes) -> char* {
    char* p = ws + off;
    off = (off + bytes + 255) & ~(size_t)255;
    return p;
  };
  __bf16* xbf    = (__bf16*)take((size_t)Bn * Sn * Dn * 2);
  __bf16* wihf   = (__bf16*)take((size_t)G4 * Dn * 2);
  __bf16* wihb   = (__bf16*)take((size_t)G4 * Dn * 2);
  __bf16* whhf   = (__bf16*)take((size_t)G4 * Hn * 2);
  __bf16* whhb   = (__bf16*)take((size_t)G4 * Hn * 2);
  __bf16* wclf16 = (__bf16*)take((size_t)16 * 2 * Hn * 2);
  __bf16* xgf    = (__bf16*)take((size_t)Sn * Bn * G4 * 2);
  __bf16* xgb    = (__bf16*)take((size_t)Sn * Bn * G4 * 2);
  __bf16* hreluf = (__bf16*)take((size_t)Bn * Sn * Hn * 2);
  __bf16* hrelub = (__bf16*)take((size_t)Bn * Sn * Hn * 2);
  __bf16* hstate = (__bf16*)take((size_t)2 * 2 * Bn * Hn * 2);
  unsigned* bar  = (unsigned*)take(256);

  // Phase 0: data/weight conversion + state/barrier init (re-runs per replay)
  prep_kernel<<<4096, 256, 0, stream>>>(
      x, WihF, WhhF, WihB, WhhB, Wclf,
      xbf, wihf, whhf, wihb, whhb, wclf16, hstate, bar);

  // Phase 1: hoisted input projection for both directions
  gemm_xg_kernel<<<dim3((Bn * Sn) / 16, 8), 256, 0, stream>>>(
      xbf, wihf, wihb, bF, bB, xgf, xgb);

  // Phase 2: persistent weight-stationary recurrent scan (both directions)
  const size_t scan_lds = 128 * 512 * 2 + 16 * 128 * 4 + 16 * 32 * 4;  // 141312 B
  scan_kernel<<<NSCAN_WG, 256, scan_lds, stream>>>(
      whhf, whhb, xgf, xgb, hreluf, hrelub, hstate, bar);

  // Phase 3: classifier
  clf_kernel<<<(Bn * Sn) / (16 * 8), 256, 0, stream>>>(
      hreluf, hrelub, wclf16, bclf, out);
}
